// PointNetSetAbstraction_28965259444562
// MI455X (gfx1250) — matrix operand order repaired and checked by reading, not compiled, not run on hardware
//
#include <hip/hip_runtime.h>

#define B_   16
#define N_   4096
#define D_   64
#define S_   1024
#define K_   32
#define CIN  67
#define KP0  96
#define M_   (B_*S_*K_)   /* 524288 rows */
#define R2_  (0.2f*0.2f)

typedef __attribute__((ext_vector_type(16))) __bf16 v16bf;
typedef __attribute__((ext_vector_type(8)))  __bf16 v8bf;
typedef __attribute__((ext_vector_type(8)))  float  v8f;

__device__ __forceinline__ __bf16 f2bf(float f) {
    unsigned u = __builtin_bit_cast(unsigned, f);
    u += 0x7fffu + ((u >> 16) & 1u);              // round-to-nearest-even
    unsigned short h = (unsigned short)(u >> 16);
    return __builtin_bit_cast(__bf16, h);
}
__device__ __forceinline__ float bf2f(__bf16 b) {
    unsigned short h = __builtin_bit_cast(unsigned short, b);
    unsigned u = ((unsigned)h) << 16;
    return __builtin_bit_cast(float, u);
}

// ---------------------------------------------------------------- FPS
// One workgroup per batch. Coordinates + running min-distance in registers
// (512 thr x 8 pts); wave-level shfl argmax, tiny LDS combine (3 syncs/iter).
__global__ __launch_bounds__(512) void fps_kernel(const float* __restrict__ xyz,
                                                  int* __restrict__ fps_idx) {
    const int T = 512, P = N_ / T;
    int b = blockIdx.x, t = threadIdx.x;
    const float* xb = xyz + (size_t)b * 3 * N_;
    float lx[P], ly[P], lz[P], ld[P];
#pragma unroll
    for (int j = 0; j < P; ++j) {
        int i = t + j * T;
        lx[j] = xb[i]; ly[j] = xb[N_ + i]; lz[j] = xb[2 * N_ + i];
        ld[j] = 1e10f;
    }
    __shared__ float wvv[16];
    __shared__ int   wva[16];
    int wv = t >> 5;
    int last = 0;
    for (int it = 0; it < S_; ++it) {
        if (t == 0) fps_idx[b * S_ + it] = last;
        float px = xb[last], py = xb[N_ + last], pz = xb[2 * N_ + last];
        float bm = -1e30f; int ba = 0;
#pragma unroll
        for (int j = 0; j < P; ++j) {
            int i = t + j * T;
            float dx = lx[j] - px, dy = ly[j] - py, dz = lz[j] - pz;
            float d = dx * dx + dy * dy + dz * dz;
            ld[j] = fminf(ld[j], d);
            if (ld[j] > bm) { bm = ld[j]; ba = i; }
        }
        // wave32 butterfly argmax (max value, min index on tie)
#pragma unroll
        for (int off = 16; off > 0; off >>= 1) {
            float ov = __shfl_xor(bm, off, 32);
            int   oa = __shfl_xor(ba, off, 32);
            if (ov > bm || (ov == bm && oa < ba)) { bm = ov; ba = oa; }
        }
        if ((t & 31) == 0) { wvv[wv] = bm; wva[wv] = ba; }
        __syncthreads();
        if (t == 0) {
            float bv = wvv[0]; int baa = wva[0];
            for (int w = 1; w < 16; ++w)
                if (wvv[w] > bv || (wvv[w] == bv && wva[w] < baa)) { bv = wvv[w]; baa = wva[w]; }
            wva[0] = baa;
        }
        __syncthreads();
        last = wva[0];
        __syncthreads();
    }
}

// ------------------------------------------------- gather sampled centers
__global__ void gather_new_xyz(const float* __restrict__ xyz,
                               const int* __restrict__ fps_idx,
                               float* __restrict__ new_xyz,   // [B,S,3]
                               float* __restrict__ out_xyz) { // [B,3,S]
    int t = blockIdx.x * blockDim.x + threadIdx.x;
    if (t >= B_ * S_) return;
    int b = t >> 10, s = t & 1023;
    int n = fps_idx[t];
    for (int c = 0; c < 3; ++c) {
        float v = xyz[((size_t)b * 3 + c) * N_ + n];
        new_xyz[(size_t)t * 3 + c] = v;
        out_xyz[((size_t)b * 3 + c) * S_ + s] = v;
    }
}

// ---------------------------------------------------------- ball query
__global__ void query_ball_k(const float* __restrict__ xyz,
                             const float* __restrict__ new_xyz,
                             int* __restrict__ ball) {         // [B,S,K]
    int t = blockIdx.x * blockDim.x + threadIdx.x;
    if (t >= B_ * S_) return;
    int b = t >> 10;
    const float* xb = xyz + (size_t)b * 3 * N_;
    float qx = new_xyz[t * 3], qy = new_xyz[t * 3 + 1], qz = new_xyz[t * 3 + 2];
    int* o = ball + (size_t)t * K_;
    int cnt = 0, first = 0;
    for (int n = 0; n < N_ && cnt < K_; ++n) {
        float dx = xb[n] - qx, dy = xb[N_ + n] - qy, dz = xb[2 * N_ + n] - qz;
        float d = dx * dx + dy * dy + dz * dz;
        if (d <= R2_) { if (cnt == 0) first = n; o[cnt++] = n; }
    }
    for (int k = cnt; k < K_; ++k) o[k] = first;
}

// ------------------------------------- points [B,D,N] f32 -> [B,N,D] bf16
__global__ void transpose_points(const float* __restrict__ pts,
                                 __bf16* __restrict__ pts_t) {
    size_t t = (size_t)blockIdx.x * blockDim.x + threadIdx.x;
    if (t >= (size_t)B_ * D_ * N_) return;
    int n = t & (N_ - 1);
    int d = (t >> 12) & (D_ - 1);
    int b = (int)(t >> 18);
    pts_t[((size_t)b * N_ + n) * D_ + d] = f2bf(pts[t]);
}

// ------------------------------------- weights f32 [O,C] -> bf16 [O,Kpad]
__global__ void convert_w(const float* __restrict__ w, __bf16* __restrict__ wb,
                          int O, int C, int Kpad) {
    int t = blockIdx.x * blockDim.x + threadIdx.x;
    if (t >= O * Kpad) return;
    int o = t / Kpad, k = t % Kpad;
    wb[t] = f2bf(k < C ? w[o * C + k] : 0.f);
}

// ----------------------- build grouped feature matrix [M, 96] bf16
__global__ __launch_bounds__(256) void build_feat0(
        const float* __restrict__ xyz, const float* __restrict__ new_xyz,
        const int* __restrict__ ball, const __bf16* __restrict__ pts_t,
        __bf16* __restrict__ feat) {
    int wid  = (blockIdx.x * blockDim.x + threadIdx.x) >> 5;
    int lane = threadIdx.x & 31;
    if (wid >= M_) return;
    int k = wid & 31, s = (wid >> 5) & 1023, b = wid >> 15;
    int bs = (b << 10) | s;
    int n = ball[(size_t)bs * K_ + k];
    __bf16* row = feat + (size_t)wid * KP0;
    const __bf16* prow = pts_t + ((size_t)b * N_ + n) * D_;
    for (int c = lane; c < KP0; c += 32) {
        __bf16 v;
        if (c < 3)           v = f2bf(xyz[((size_t)b * 3 + c) * N_ + n] - new_xyz[(size_t)bs * 3 + c]);
        else if (c < 3 + D_) v = prow[c - 3];
        else                 v = f2bf(0.f);
        row[c] = v;
    }
}

// =====================================================================
// Fused GEMM: optional BN+ReLU applied to A fragments while loading (FUSE),
// register-resident B fragments reused over MT m-tiles, bias in epilogue,
// per-channel sum/sumsq reduced in LDS and atomically added to `stats`.
// Wave = 16*MT rows x (NTW*16) cols; blockIdx.y selects the column group.
template <int KPAD, int O, int NTW, int MT, bool FUSE>
__global__ __launch_bounds__(256) void gemm_bf16_fused(
        const __bf16* __restrict__ A, const __bf16* __restrict__ W,
        const float* __restrict__ bias, const float* __restrict__ abPrev,
        __bf16* __restrict__ out, float* __restrict__ stats, int M) {
    constexpr int KT = KPAD / 32;
    constexpr int NC = NTW * 16;
    int lane = threadIdx.x & 31;
    int wv   = threadIdx.x >> 5;
    int hi = lane >> 4, lo = lane & 15;
    int n0 = blockIdx.y * NC;
    size_t m0 = ((size_t)blockIdx.x * 8 + wv) * (16 * MT);

    // ---- weights -> registers (reused for all MT m-tiles)
    v16bf Bf[KT][NTW];
#pragma unroll
    for (int kt = 0; kt < KT; ++kt)
#pragma unroll
        for (int nt = 0; nt < NTW; ++nt) {
            const __bf16* wr = W + (size_t)(n0 + nt * 16 + lo) * KPAD + kt * 32 + hi * 8;
            v8bf w0v = *reinterpret_cast<const v8bf*>(wr);
            v8bf w1v = *reinterpret_cast<const v8bf*>(wr + 16);
#pragma unroll
            for (int j = 0; j < 8; ++j) { Bf[kt][nt][j] = w0v[j]; Bf[kt][nt][8 + j] = w1v[j]; }
        }

    __shared__ float bsum[NC], bsq[NC];
    for (int i = threadIdx.x; i < NC; i += 256) { bsum[i] = 0.f; bsq[i] = 0.f; }
    __syncthreads();

    float pS[NTW] = {0.f}, pQ[NTW] = {0.f};

#pragma unroll
    for (int mt = 0; mt < MT; ++mt) {
        const __bf16* arow = A + (m0 + mt * 16 + lo) * KPAD;
        v16bf Af[KT];
#pragma unroll
        for (int kt = 0; kt < KT; ++kt) {
            int k0 = kt * 32 + hi * 8;
            v8bf a0 = *reinterpret_cast<const v8bf*>(arow + k0);
            v8bf a1 = *reinterpret_cast<const v8bf*>(arow + k0 + 16);
            if constexpr (FUSE) {
#pragma unroll
                for (int j = 0; j < 8; ++j) {
                    int ka = k0 + j, kb = k0 + 16 + j;
                    Af[kt][j]     = f2bf(fmaxf(bf2f(a0[j]) * abPrev[ka] + abPrev[KPAD + ka], 0.f));
                    Af[kt][8 + j] = f2bf(fmaxf(bf2f(a1[j]) * abPrev[kb] + abPrev[KPAD + kb], 0.f));
                }
            } else {
#pragma unroll
                for (int j = 0; j < 8; ++j) { Af[kt][j] = a0[j]; Af[kt][8 + j] = a1[j]; }
            }
        }
        v8f acc[NTW];
#pragma unroll
        for (int nt = 0; nt < NTW; ++nt) acc[nt] = v8f{};
#pragma unroll
        for (int kt = 0; kt < KT; ++kt)
#pragma unroll
            for (int nt = 0; nt < NTW; ++nt)
                acc[nt] = __builtin_amdgcn_wmma_f32_16x16x32_bf16(
                    false, Af[kt], false, Bf[kt][nt], (short)0, acc[nt], false, false);
        // ---- epilogue: bias add, bf16 store, stats partials
#pragma unroll
        for (int nt = 0; nt < NTW; ++nt) {
            int n = n0 + nt * 16 + lo;
            float bv = bias[n];
            size_t mrow = m0 + mt * 16 + 8 * hi;
#pragma unroll
            for (int i = 0; i < 8; ++i) {
                float v = acc[nt][i] + bv;
                out[(mrow + i) * O + n] = f2bf(v);
                pS[nt] += v; pQ[nt] += v * v;
            }
        }
    }
#pragma unroll
    for (int nt = 0; nt < NTW; ++nt) {
        atomicAdd(&bsum[nt * 16 + lo], pS[nt]);
        atomicAdd(&bsq [nt * 16 + lo], pQ[nt]);
    }
    __syncthreads();
    for (int i = threadIdx.x; i < NC; i += 256) {
        atomicAdd(&stats[n0 + i],     bsum[i]);
        atomicAdd(&stats[O + n0 + i], bsq[i]);
    }
}

// --------------------------------------------------------------- BN glue
__global__ void zero_f32(float* p, int n) {
    int t = blockIdx.x * blockDim.x + threadIdx.x;
    if (t < n) p[t] = 0.f;
}

__global__ void bn_finalize(const float* __restrict__ sums,
                            const float* __restrict__ g,
                            const float* __restrict__ be,
                            float* __restrict__ ab, int O, float invM) {
    int c = threadIdx.x;
    if (c >= O) return;
    float mu  = sums[c] * invM;
    float var = sums[O + c] * invM - mu * mu;
    float A = g[c] * rsqrtf(var + 1e-5f);
    ab[c] = A;
    ab[O + c] = be[c] - mu * A;
}

// --------------------- BN+ReLU of layer2 fused into max-pool + transpose
__global__ void maxpool_bn(const __bf16* __restrict__ feat,   // [M,128] raw
                           const float* __restrict__ ab,       // [256]
                           float* __restrict__ out) {          // [B,128,S]
    int t = blockIdx.x * blockDim.x + threadIdx.x;
    if (t >= B_ * S_ * 128) return;
    int o = t & 127;
    int s = (t >> 7) & 1023;
    int b = t >> 17;
    float sc = ab[o], sh = ab[128 + o];
    size_t base = (((size_t)b * S_ + s) * K_) * 128 + o;
    float mx = fmaxf(bf2f(feat[base]) * sc + sh, 0.f);
    for (int k = 1; k < K_; ++k)
        mx = fmaxf(mx, fmaxf(bf2f(feat[base + (size_t)k * 128]) * sc + sh, 0.f));
    out[((size_t)b * 128 + o) * S_ + s] = mx;
}

// =====================================================================
extern "C" void kernel_launch(void* const* d_in, const int* in_sizes, int n_in,
                              void* d_out, int out_size, void* d_ws, size_t ws_size,
                              hipStream_t stream) {
    (void)in_sizes; (void)n_in; (void)out_size; (void)ws_size;
    const float* xyz = (const float*)d_in[0];
    const float* pts = (const float*)d_in[1];
    const float* w0 = (const float*)d_in[2],  *b0 = (const float*)d_in[3];
    const float* g0 = (const float*)d_in[4],  *be0 = (const float*)d_in[5];
    const float* w1 = (const float*)d_in[6],  *b1 = (const float*)d_in[7];
    const float* g1 = (const float*)d_in[8],  *be1 = (const float*)d_in[9];
    const float* w2 = (const float*)d_in[10], *b2 = (const float*)d_in[11];
    const float* g2 = (const float*)d_in[12], *be2 = (const float*)d_in[13];

    float* out_xyz = (float*)d_out;            // [B,3,S]
    float* out_pts = out_xyz + B_ * 3 * S_;    // [B,128,S]

    char* ws = (char*)d_ws;
    size_t off = 0;
    auto take = [&](size_t bytes) -> char* {
        char* p = ws + off;
        off = (off + bytes + 255) & ~(size_t)255;
        return p;
    };
    int*    fps_idx = (int*)   take((size_t)B_ * S_ * 4);
    float*  new_xyz = (float*) take((size_t)B_ * S_ * 3 * 4);
    int*    ball    = (int*)   take((size_t)B_ * S_ * K_ * 4);
    __bf16* pts_t   = (__bf16*)take((size_t)B_ * N_ * D_ * 2);
    __bf16* wb0     = (__bf16*)take((size_t)64 * KP0 * 2);
    __bf16* wb1     = (__bf16*)take((size_t)64 * 64 * 2);
    __bf16* wb2     = (__bf16*)take((size_t)128 * 64 * 2);
    float*  stats   = (float*) take(768 * 4);   // 3 layers x [2*O]
    float*  abbuf   = (float*) take(512 * 4);   // ab0[128] ab1[128] ab2[256]
    __bf16* bufA    = (__bf16*)take((size_t)M_ * KP0 * 2);   // feat0 / layer1 out
    __bf16* bufB    = (__bf16*)take((size_t)M_ * 128 * 2);   // layer0 out / layer2 out

    float* stats0 = stats,       *stats1 = stats + 256, *stats2 = stats + 512;
    float* ab0    = abbuf,       *ab1    = abbuf + 128, *ab2    = abbuf + 256;

    // ---- sampling / grouping / precomputation
    fps_kernel<<<B_, 512, 0, stream>>>(xyz, fps_idx);
    gather_new_xyz<<<(B_ * S_ + 255) / 256, 256, 0, stream>>>(xyz, fps_idx, new_xyz, out_xyz);
    query_ball_k<<<(B_ * S_ + 255) / 256, 256, 0, stream>>>(xyz, new_xyz, ball);
    transpose_points<<<(B_ * D_ * N_) / 256, 256, 0, stream>>>(pts, pts_t);
    convert_w<<<(64 * KP0 + 255) / 256, 256, 0, stream>>>(w0, wb0, 64, CIN, KP0);
    convert_w<<<(64 * 64 + 255) / 256, 256, 0, stream>>>(w1, wb1, 64, 64, 64);
    convert_w<<<(128 * 64 + 255) / 256, 256, 0, stream>>>(w2, wb2, 128, 64, 64);
    zero_f32<<<3, 256, 0, stream>>>(stats, 768);
    build_feat0<<<M_ / 8, 256, 0, stream>>>(xyz, new_xyz, ball, pts_t, bufA);

    const float invM = 1.0f / (float)M_;
    // block covers 8 waves * 64 rows = 512 rows -> 1024 blocks exactly
    const int GX = M_ / 512;

    // ---- layer 0: [M,96] x [96,64], raw input
    gemm_bf16_fused<KP0, 64, 4, 4, false>
        <<<dim3(GX, 1), 256, 0, stream>>>(bufA, wb0, b0, ab0, bufB, stats0, M_);
    bn_finalize<<<1, 128, 0, stream>>>(stats0, g0, be0, ab0, 64, invM);

    // ---- layer 1: [M,64] x [64,64], BN0+ReLU fused into A load
    gemm_bf16_fused<64, 64, 4, 4, true>
        <<<dim3(GX, 1), 256, 0, stream>>>(bufB, wb1, b1, ab0, bufA, stats1, M_);
    bn_finalize<<<1, 128, 0, stream>>>(stats1, g1, be1, ab1, 64, invM);

    // ---- layer 2: [M,64] x [64,128], BN1+ReLU fused; N split over grid.y
    gemm_bf16_fused<64, 128, 4, 4, true>
        <<<dim3(GX, 2), 256, 0, stream>>>(bufA, wb2, b2, ab1, bufB, stats2, M_);
    bn_finalize<<<1, 128, 0, stream>>>(stats2, g2, be2, ab2, 128, invM);

    // ---- BN2+ReLU + max over K + transpose to [B,128,S]
    maxpool_bn<<<(B_ * S_ * 128) / 256, 256, 0, stream>>>(bufB, ab2, out_pts);
}
